// Net_19576460935594
// MI455X (gfx1250) — matrix operand order repaired and checked by reading, not compiled, not run on hardware
//
#include <hip/hip_runtime.h>

#define D 128
#define EPSV 1e-5f

typedef __attribute__((ext_vector_type(16))) __bf16 v16bf;
typedef __attribute__((ext_vector_type(8)))  float  v8f;

union BF16x16 { v16bf v; __bf16 bs[16]; unsigned short us[16]; uint4 q[2]; };

// native hardware bf16 convert (v_cvt_pk_bf16_f32 on gfx1250)
__device__ __forceinline__ unsigned short f2bf(float f) {
    union { __bf16 b; unsigned short u; } cv;
    cv.b = (__bf16)f;
    return cv.u;
}

__device__ __forceinline__ v8f wmma_bf16(v16bf a, v16bf b, v8f c) {
    return __builtin_amdgcn_wmma_f32_16x16x32_bf16(
        false, a, false, b, (short)0, c, false, false);
}

// B fragment: pre-packed so each lane reads 16 contiguous bf16 (two b128 loads)
__device__ __forceinline__ v16bf load_bfrag(const unsigned short* pack, int cb, int nt, int lane) {
    const uint4* p = (const uint4*)(pack + (((cb * 8 + nt) * 32 + lane) << 4));
    BF16x16 u; u.q[0] = p[0]; u.q[1] = p[1];
    return u.v;
}

// A fragment from a row-major bf16 row: two 8-element runs (K=kb.., K=kb+16..)
__device__ __forceinline__ v16bf load_afrag_bf(const unsigned short* row, int kb) {
    BF16x16 u;
    u.q[0] = *(const uint4*)(row + kb);
    u.q[1] = *(const uint4*)(row + kb + 16);
    return u.v;
}

// chunk-block bases (edge-phase blocks first so they stage contiguously)
#define CB_DIST  0      // 4 blocks
#define CB_CTX1  4      // 12 blocks
#define CB_CTX2  16     // 4 blocks
#define CB_QUERY 20     // 4 blocks
#define CB_AGT   24     // 4 blocks
#define CB_LIN   28     // 4 blocks
#define EDGE_PACK_BYTES (20 * 4096 * 2)        // 160 KB: cb 0..19
#define EDGE_PACK_UNITS (EDGE_PACK_BYTES / 16) // 16B async units

// ---------------- weight pack: fp32 row-major -> bf16 WMMA-B fragment order ----
__global__ __launch_bounds__(256)
void pack_weights_kernel(const float* dist_w1, const float* query_w,
                         const float* ctx_w1, const float* ctx_w2,
                         const float* agt_w, const float* lin_w,
                         unsigned short* pack) {
    int cb = blockIdx.x;           // 0..31 chunk-blocks (K-chunks of 32)
    const float* W; int kc;
    if      (cb < 4)  { W = dist_w1;  kc = cb;      }
    else if (cb < 16) { W = ctx_w1;   kc = cb - 4;  }
    else if (cb < 20) { W = ctx_w2;   kc = cb - 16; }
    else if (cb < 24) { W = query_w;  kc = cb - 20; }
    else if (cb < 28) { W = agt_w;    kc = cb - 24; }
    else              { W = lin_w;    kc = cb - 28; }
    for (int e = threadIdx.x; e < 4096; e += blockDim.x) {
        int nt   = e >> 9;
        int lane = (e >> 4) & 31;
        int j2   = e & 15;
        int krow = kc * 32 + ((j2 >> 3) << 4) + ((lane >> 4) << 3) + (j2 & 7);  // B: lanes 16-31 hold K+8 (runs K..K+7, K+16..K+23)
        int col  = nt * 16 + (lane & 15);
        pack[cb * 4096 + e] = f2bf(W[krow * D + col]);
    }
}

// ---------------- ctx cast fp32 -> bf16 ----------------
__global__ __launch_bounds__(256)
void cast_bf16_kernel(const float* src, unsigned short* dst, int total) {
    for (int i = blockIdx.x * blockDim.x + threadIdx.x; i < total;
         i += gridDim.x * blockDim.x)
        dst[i] = f2bf(src[i]);
}

// ---------------- agent pre: Q = relu(gn(agts@Wq)), Aacc = agts@Wa ----------------
__global__ __launch_bounds__(128)
void agent_pre_kernel(const float* __restrict__ agts,
                      const unsigned short* __restrict__ pack,
                      const float* __restrict__ qg, const float* __restrict__ qb,
                      unsigned short* __restrict__ Qbf,
                      float* __restrict__ Aacc, int N) {
    int lane = threadIdx.x & 31, wid = threadIdx.x >> 5;
    int h = lane >> 4, l = lane & 15;
    int ntiles = (N + 15) >> 4;
    for (int tile = blockIdx.x * 4 + wid; tile < ntiles; tile += gridDim.x * 4) {
        int grow = tile * 16 + l; if (grow >= N) grow = N - 1;
        const float* row = agts + (size_t)grow * D;
        v16bf aA[4];
#pragma unroll
        for (int kc = 0; kc < 4; kc++) {
            BF16x16 u;
#pragma unroll
            for (int part = 0; part < 2; part++) {
                const float4* p = (const float4*)(row + kc * 32 + part * 16 + h * 8);
                float4 a0 = p[0], a1 = p[1];
                int b = part * 8;
                u.bs[b+0]=(__bf16)a0.x; u.bs[b+1]=(__bf16)a0.y;
                u.bs[b+2]=(__bf16)a0.z; u.bs[b+3]=(__bf16)a0.w;
                u.bs[b+4]=(__bf16)a1.x; u.bs[b+5]=(__bf16)a1.y;
                u.bs[b+6]=(__bf16)a1.z; u.bs[b+7]=(__bf16)a1.w;
            }
            aA[kc] = u.v;
        }
        v8f acc[8];
        // ---- query GEMM ----
#pragma unroll
        for (int nt = 0; nt < 8; nt++) {
            v8f c = {0,0,0,0,0,0,0,0};
#pragma unroll
            for (int kc = 0; kc < 4; kc++)
                c = wmma_bf16(aA[kc], load_bfrag(pack, CB_QUERY + kc, nt, lane), c);
            acc[nt] = c;
        }
        float garr[8], barr[8];
#pragma unroll
        for (int nt = 0; nt < 8; nt++) { garr[nt] = qg[nt*16 + l]; barr[nt] = qb[nt*16 + l]; }
#pragma unroll
        for (int v = 0; v < 8; v++) {
            float s = 0.f, s2 = 0.f;
#pragma unroll
            for (int nt = 0; nt < 8; nt++) { float x = acc[nt][v]; s += x; s2 += x*x; }
#pragma unroll
            for (int m = 1; m < 16; m <<= 1) { s += __shfl_xor(s, m, 32); s2 += __shfl_xor(s2, m, 32); }
            float mean = s * (1.f/128.f);
            float inv  = rsqrtf(s2 * (1.f/128.f) - mean*mean + EPSV);
            int grow2 = tile * 16 + v + 8 * h;
            if (grow2 < N) {
#pragma unroll
                for (int nt = 0; nt < 8; nt++) {
                    float val = fmaxf((acc[nt][v] - mean) * inv * garr[nt] + barr[nt], 0.f);
                    Qbf[(size_t)grow2 * D + nt*16 + l] = f2bf(val);
                }
            }
        }
        // ---- agt GEMM -> fp32 accumulator ----
#pragma unroll
        for (int nt = 0; nt < 8; nt++) {
            v8f c = {0,0,0,0,0,0,0,0};
#pragma unroll
            for (int kc = 0; kc < 4; kc++)
                c = wmma_bf16(aA[kc], load_bfrag(pack, CB_AGT + kc, nt, lane), c);
            acc[nt] = c;
        }
#pragma unroll
        for (int v = 0; v < 8; v++) {
            int grow2 = tile * 16 + v + 8 * h;
            if (grow2 < N) {
#pragma unroll
                for (int nt = 0; nt < 8; nt++)
                    Aacc[(size_t)grow2 * D + nt*16 + l] = acc[nt][v];
            }
        }
    }
}

// ---------------- edge kernel (weights async-staged in LDS) ----------------
__global__ __launch_bounds__(256)
void edge_kernel(const float* __restrict__ agt_ctrs, const float* __restrict__ ctx_ctrs,
                 const int* __restrict__ hi, const int* __restrict__ wi,
                 const float* __restrict__ dw0, const float* __restrict__ db0,
                 const float* __restrict__ dg1, const float* __restrict__ dbt1,
                 const float* __restrict__ cg1, const float* __restrict__ cb1,
                 const unsigned short* __restrict__ pack,
                 const unsigned short* __restrict__ Qbf,
                 const unsigned short* __restrict__ Cbf,
                 float* __restrict__ Aacc, int E) {
    extern __shared__ __align__(16) unsigned char smem[];
    unsigned short* lpack = (unsigned short*)smem;                       // 160 KB
    int lane = threadIdx.x & 31, wid = threadIdx.x >> 5;
    int h = lane >> 4, l = lane & 15;
    unsigned short* stage = (unsigned short*)(smem + EDGE_PACK_BYTES + wid * (16 * D * 2));

    // ---- async-stage the edge weight pack into LDS (ASYNCcnt path) ----
    {
        unsigned ldsBase = (unsigned)(size_t)(void*)lpack;   // LDS aperture: low 32 bits = LDS offset
        const unsigned char* gsrc = (const unsigned char*)pack;
        for (int i = threadIdx.x; i < EDGE_PACK_UNITS; i += 256) {
            unsigned dst = ldsBase + (unsigned)i * 16u;
            const unsigned char* src = gsrc + (size_t)i * 16u;
            asm volatile("global_load_async_to_lds_b128 %0, %1, off"
                         :: "v"(dst), "v"(src) : "memory");
        }
        asm volatile("s_wait_asynccnt 0" ::: "memory");
        __syncthreads();
    }

    int etiles = (E + 15) >> 4;
    for (int tile = blockIdx.x * 8 + wid; tile < etiles; tile += gridDim.x * 8) {
        int e = tile * 16 + l; if (e >= E) e = E - 1;
        int hrow = hi[e], wrow = wi[e];
        float dx = agt_ctrs[2*hrow]   - ctx_ctrs[2*wrow];
        float dy = agt_ctrs[2*hrow+1] - ctx_ctrs[2*wrow+1];

        // h0 = relu([dx,dy]@W0 + b0), built directly in A-fragment registers
        v16bf aH[4];
#pragma unroll
        for (int kc = 0; kc < 4; kc++) {
            BF16x16 u;
#pragma unroll
            for (int part = 0; part < 2; part++) {
                int kb = kc * 32 + part * 16 + h * 8;
#pragma unroll
                for (int j = 0; j < 8; j++) {
                    int k = kb + j;
                    float v = fmaf(dx, dw0[k], fmaf(dy, dw0[D + k], db0[k]));
                    u.bs[part*8 + j] = (__bf16)fmaxf(v, 0.f);
                }
            }
            aH[kc] = u.v;
        }

        v8f acc[8];
        // GEMM1: h0 @ dist_w1 (B frags from LDS)
#pragma unroll
        for (int nt = 0; nt < 8; nt++) {
            v8f c = {0,0,0,0,0,0,0,0};
#pragma unroll
            for (int kc = 0; kc < 4; kc++)
                c = wmma_bf16(aH[kc], load_bfrag(lpack, CB_DIST + kc, nt, lane), c);
            acc[nt] = c;
        }
        // GN + ReLU (C layout) -> LDS row-major bf16
        float garr[8], barr[8];
#pragma unroll
        for (int nt = 0; nt < 8; nt++) { garr[nt] = dg1[nt*16 + l]; barr[nt] = dbt1[nt*16 + l]; }
#pragma unroll
        for (int v = 0; v < 8; v++) {
            float s = 0.f, s2 = 0.f;
#pragma unroll
            for (int nt = 0; nt < 8; nt++) { float x = acc[nt][v]; s += x; s2 += x*x; }
#pragma unroll
            for (int m = 1; m < 16; m <<= 1) { s += __shfl_xor(s, m, 32); s2 += __shfl_xor(s2, m, 32); }
            float mean = s * (1.f/128.f);
            float inv  = rsqrtf(s2 * (1.f/128.f) - mean*mean + EPSV);
            int r = v + 8 * h;
#pragma unroll
            for (int nt = 0; nt < 8; nt++) {
                float val = fmaxf((acc[nt][v] - mean) * inv * garr[nt] + barr[nt], 0.f);
                stage[r * D + nt*16 + l] = f2bf(val);
            }
        }
        asm volatile("s_wait_dscnt 0" ::: "memory");

        // reload as A fragments; gather Q[hi], ctx_bf[wi]
        v16bf aD2[4], aQ[4], aC[4];
        const unsigned short* qrow = Qbf + (size_t)hrow * D;
        const unsigned short* crow = Cbf + (size_t)wrow * D;
#pragma unroll
        for (int kc = 0; kc < 4; kc++) {
            int kb = kc * 32 + 8 * h;
            aD2[kc] = load_afrag_bf(stage + l * D, kb);
            aQ[kc]  = load_afrag_bf(qrow, kb);
            aC[kc]  = load_afrag_bf(crow, kb);
        }
        // GEMM2: [d|q|ctx] @ ctx_w1  (12 K-chunks)
#pragma unroll
        for (int nt = 0; nt < 8; nt++) {
            v8f c = {0,0,0,0,0,0,0,0};
#pragma unroll
            for (int kc = 0; kc < 12; kc++) {
                v16bf a = (kc < 4) ? aD2[kc] : (kc < 8) ? aQ[kc-4] : aC[kc-8];
                c = wmma_bf16(a, load_bfrag(lpack, CB_CTX1 + kc, nt, lane), c);
            }
            acc[nt] = c;
        }
        // GN + ReLU -> LDS
#pragma unroll
        for (int nt = 0; nt < 8; nt++) { garr[nt] = cg1[nt*16 + l]; barr[nt] = cb1[nt*16 + l]; }
#pragma unroll
        for (int v = 0; v < 8; v++) {
            float s = 0.f, s2 = 0.f;
#pragma unroll
            for (int nt = 0; nt < 8; nt++) { float x = acc[nt][v]; s += x; s2 += x*x; }
#pragma unroll
            for (int m = 1; m < 16; m <<= 1) { s += __shfl_xor(s, m, 32); s2 += __shfl_xor(s2, m, 32); }
            float mean = s * (1.f/128.f);
            float inv  = rsqrtf(s2 * (1.f/128.f) - mean*mean + EPSV);
            int r = v + 8 * h;
#pragma unroll
            for (int nt = 0; nt < 8; nt++) {
                float val = fmaxf((acc[nt][v] - mean) * inv * garr[nt] + barr[nt], 0.f);
                stage[r * D + nt*16 + l] = f2bf(val);
            }
        }
        asm volatile("s_wait_dscnt 0" ::: "memory");

        v16bf aG[4];
#pragma unroll
        for (int kc = 0; kc < 4; kc++)
            aG[kc] = load_afrag_bf(stage + l * D, kc * 32 + 8 * h);
        // GEMM3: @ ctx_w2
#pragma unroll
        for (int nt = 0; nt < 8; nt++) {
            v8f c = {0,0,0,0,0,0,0,0};
#pragma unroll
            for (int kc = 0; kc < 4; kc++)
                c = wmma_bf16(aG[kc], load_bfrag(lpack, CB_CTX2 + kc, nt, lane), c);
            acc[nt] = c;
        }
        // scatter-add into agent accumulator (L2-resident fp32 atomics)
#pragma unroll
        for (int v = 0; v < 8; v++) {
            int er = tile * 16 + v + 8 * h;
            if (er < E) {
                int dst = hi[er];
                float* base = Aacc + (size_t)dst * D + l;
#pragma unroll
                for (int nt = 0; nt < 8; nt++)
                    atomicAdd(base + nt * 16, acc[nt][v]);
            }
        }
    }
}

// ---------------- agent post: relu(gn(Aacc)) @ lin_w -> gn -> +res -> relu ----------------
__global__ __launch_bounds__(128)
void agent_post_kernel(const float* __restrict__ agts,
                       const float* __restrict__ Aacc,
                       const float* __restrict__ ng, const float* __restrict__ nb,
                       const float* __restrict__ lg, const float* __restrict__ lb,
                       const unsigned short* __restrict__ pack,
                       float* __restrict__ out, int N) {
    int lane = threadIdx.x & 31, wid = threadIdx.x >> 5;
    int h = lane >> 4, l = lane & 15;
    int ntiles = (N + 15) >> 4;
    for (int tile = blockIdx.x * 4 + wid; tile < ntiles; tile += gridDim.x * 4) {
        int grow = tile * 16 + l; if (grow >= N) grow = N - 1;
        const float* arow = Aacc + (size_t)grow * D;
        float xv[64];
        float s = 0.f, s2 = 0.f;
#pragma unroll
        for (int kc = 0; kc < 4; kc++) {
#pragma unroll
            for (int part = 0; part < 2; part++) {
                const float4* p = (const float4*)(arow + kc * 32 + part * 16 + h * 8);
                float4 a0 = p[0], a1 = p[1];
                int b = kc * 16 + part * 8;
                xv[b+0]=a0.x; xv[b+1]=a0.y; xv[b+2]=a0.z; xv[b+3]=a0.w;
                xv[b+4]=a1.x; xv[b+5]=a1.y; xv[b+6]=a1.z; xv[b+7]=a1.w;
#pragma unroll
                for (int j = 0; j < 8; j++) { float x = xv[b+j]; s += x; s2 += x*x; }
            }
        }
        s  += __shfl_xor(s, 16, 32);
        s2 += __shfl_xor(s2, 16, 32);
        float mean = s * (1.f/128.f);
        float inv  = rsqrtf(s2 * (1.f/128.f) - mean*mean + EPSV);
        v16bf aN[4];
#pragma unroll
        for (int kc = 0; kc < 4; kc++) {
            BF16x16 u;
#pragma unroll
            for (int part = 0; part < 2; part++) {
#pragma unroll
                for (int j = 0; j < 8; j++) {
                    int idx = kc * 32 + part * 16 + h * 8 + j;
                    float val = (xv[kc*16 + part*8 + j] - mean) * inv * ng[idx] + nb[idx];
                    u.bs[part*8 + j] = (__bf16)fmaxf(val, 0.f);
                }
            }
            aN[kc] = u.v;
        }
        v8f acc[8];
#pragma unroll
        for (int nt = 0; nt < 8; nt++) {
            v8f c = {0,0,0,0,0,0,0,0};
#pragma unroll
            for (int kc = 0; kc < 4; kc++)
                c = wmma_bf16(aN[kc], load_bfrag(pack, CB_LIN + kc, nt, lane), c);
            acc[nt] = c;
        }
        float garr[8], barr[8];
#pragma unroll
        for (int nt = 0; nt < 8; nt++) { garr[nt] = lg[nt*16 + l]; barr[nt] = lb[nt*16 + l]; }
#pragma unroll
        for (int v = 0; v < 8; v++) {
            float ss = 0.f, ss2 = 0.f;
#pragma unroll
            for (int nt = 0; nt < 8; nt++) { float x = acc[nt][v]; ss += x; ss2 += x*x; }
#pragma unroll
            for (int m = 1; m < 16; m <<= 1) { ss += __shfl_xor(ss, m, 32); ss2 += __shfl_xor(ss2, m, 32); }
            float mu  = ss * (1.f/128.f);
            float iv  = rsqrtf(ss2 * (1.f/128.f) - mu*mu + EPSV);
            int grow2 = tile * 16 + v + 8 * h;
            if (grow2 < N) {
#pragma unroll
                for (int nt = 0; nt < 8; nt++) {
                    int col = nt * 16 + l;
                    float val = (acc[nt][v] - mu) * iv * garr[nt] + barr[nt];
                    float res = agts[(size_t)grow2 * D + col];
                    out[(size_t)grow2 * D + col] = fmaxf(val + res, 0.f);
                }
            }
        }
    }
}

extern "C" void kernel_launch(void* const* d_in, const int* in_sizes, int n_in,
                              void* d_out, int out_size, void* d_ws, size_t ws_size,
                              hipStream_t stream) {
    const float* agts     = (const float*)d_in[0];
    const float* ctx      = (const float*)d_in[1];
    const float* agt_ctrs = (const float*)d_in[2];
    const float* ctx_ctrs = (const float*)d_in[3];
    const int*   hi       = (const int*)d_in[4];
    const int*   wi       = (const int*)d_in[5];
    const float* dist_w0  = (const float*)d_in[6];
    const float* dist_b0  = (const float*)d_in[7];
    const float* dist_w1  = (const float*)d_in[8];
    const float* dist_g1  = (const float*)d_in[9];
    const float* dist_bt1 = (const float*)d_in[10];
    const float* query_w  = (const float*)d_in[11];
    const float* query_g  = (const float*)d_in[12];
    const float* query_b  = (const float*)d_in[13];
    const float* ctx_w1   = (const float*)d_in[14];
    const float* ctx_g1   = (const float*)d_in[15];
    const float* ctx_b1   = (const float*)d_in[16];
    const float* ctx_w2   = (const float*)d_in[17];
    const float* agt_w    = (const float*)d_in[18];
    const float* norm_g   = (const float*)d_in[19];
    const float* norm_b   = (const float*)d_in[20];
    const float* lin_w    = (const float*)d_in[21];
    const float* lin_g    = (const float*)d_in[22];
    const float* lin_b    = (const float*)d_in[23];

    int N = in_sizes[0] / D;
    int M = in_sizes[1] / D;
    int E = in_sizes[4];

    // workspace carve (all 256B aligned)
    char* ws = (char*)d_ws;
    size_t off = 0;
    unsigned short* pack = (unsigned short*)(ws + off); off += 32 * 4096 * 2;            // 256 KB
    unsigned short* Qbf  = (unsigned short*)(ws + off); off += (size_t)N * D * 2;        // 25.6 MB
    off = (off + 255) & ~(size_t)255;
    unsigned short* Cbf  = (unsigned short*)(ws + off); off += (size_t)M * D * 2;        // 25.6 MB
    off = (off + 255) & ~(size_t)255;
    float* Aacc          = (float*)(ws + off);                                            // 51.2 MB

    int ntiles = (N + 15) >> 4;
    int etiles = (E + 15) >> 4;
    int ablocks = (ntiles + 3) / 4;
    int eblocks = (etiles + 7) / 8;
    if (eblocks > 1024) eblocks = 1024;
    size_t edge_lds = EDGE_PACK_BYTES + 8 * (16 * D * 2);   // 160KB pack + 32KB stages

    (void)hipFuncSetAttribute((const void*)edge_kernel,
                              hipFuncAttributeMaxDynamicSharedMemorySize,
                              (int)edge_lds);

    pack_weights_kernel<<<32, 256, 0, stream>>>(dist_w1, query_w, ctx_w1, ctx_w2,
                                                agt_w, lin_w, pack);
    cast_bf16_kernel<<<2048, 256, 0, stream>>>(ctx, Cbf, M * D);
    agent_pre_kernel<<<ablocks, 128, 0, stream>>>(agts, pack, query_g, query_b,
                                                  Qbf, Aacc, N);
    edge_kernel<<<eblocks, 256, edge_lds, stream>>>(agt_ctrs, ctx_ctrs, hi, wi,
                                                    dist_w0, dist_b0, dist_g1, dist_bt1,
                                                    ctx_g1, ctx_b1, pack, Qbf, Cbf, Aacc, E);
    agent_post_kernel<<<ablocks, 128, 0, stream>>>(agts, Aacc, norm_g, norm_b,
                                                   lin_g, lin_b, pack,
                                                   (float*)d_out, N);
}
